// MoeExpertsEP_89292370084489
// MI455X (gfx1250) — matrix operand order repaired and checked by reading, not compiled, not run on hardware
//
#include <hip/hip_runtime.h>

// ---------------------------------------------------------------------------
// MoE experts (gate_up -> silu*up -> down), bf16 WMMA path for gfx1250.
// BK=64 K-chunks, async global->LDS for the bf16 A-tile copy in kernel 2.
// ---------------------------------------------------------------------------

#define TOKENS 2048
#define HID    1024
#define INTER  1408
#define NEXP   8

#define BM   128   // tokens per block tile
#define BN   64    // output columns per block tile
#define BK   64    // K depth per LDS stage (two bf16 WMMA K-groups)
#define LDSP 72    // padded LDS row stride in ushorts (144B -> 16B aligned)

typedef unsigned int   u32;
typedef unsigned short u16;
typedef __attribute__((ext_vector_type(4)))  u32    u32x4;
typedef __attribute__((ext_vector_type(4)))  int    i32x4;
typedef __attribute__((ext_vector_type(4)))  float  f32x4;
typedef __attribute__((ext_vector_type(16))) __bf16 v16bf;
typedef __attribute__((ext_vector_type(8)))  float  v8f;

// address-space qualified pointee types for the async builtin
typedef __attribute__((address_space(1))) i32x4 gas_i32x4;   // global (AS1)
typedef __attribute__((address_space(3))) i32x4 las_i32x4;   // LDS (AS3)

union Frag { v16bf v; u32x4 q[2]; };

#if defined(__has_builtin)
#if __has_builtin(__builtin_amdgcn_global_load_async_to_lds_b128) && \
    __has_builtin(__builtin_amdgcn_s_wait_asynccnt)
#define HAVE_ASYNC_LDS 1
#endif
#endif

// pack two f32 -> two bf16 in one dword (lo in [15:0], hi in [31:16])
// (cvt_pk_bf16_f32 builtin not available for gfx1250 on this toolchain —
//  probe-confirmed round 3 — so use a cheap round-to-nearest shift/merge.)
__device__ __forceinline__ u32 pk2(float lo, float hi) {
  union { float f; u32 u; } a, b;
  a.f = lo; b.f = hi;
  u32 al = a.u + 0x8000u;           // round-to-nearest (ties away)
  u32 bh = b.u + 0x8000u;
  return (al >> 16) | (bh & 0xffff0000u);
}
__device__ __forceinline__ u16 f2bf(float f) {
  return (u16)(pk2(f, 0.0f) & 0xffffu);
}
__device__ __forceinline__ u32x4 pk8(f32x4 a, f32x4 b) {
  u32x4 r;
  r.x = pk2(a.x, a.y); r.y = pk2(a.z, a.w);
  r.z = pk2(b.x, b.y); r.w = pk2(b.z, b.w);
  return r;
}

__device__ __forceinline__ int find_expert(const int* __restrict__ counts, int tok0) {
  int e = 0, start = 0;
  #pragma unroll
  for (int i = 0; i < NEXP - 1; ++i) {
    int c = counts[i];
    if (tok0 >= start + c) { start += c; e = i + 1; } else { break; }
  }
  return e;
}

// ---------------------------------------------------------------------------
// Kernel 1: h[T, INTER] (bf16) = silu(X @ Wg^T) * (X @ Wu^T)   per expert seg
// ---------------------------------------------------------------------------
__global__ __launch_bounds__(256) void moe_gateup_silu(
    const float* __restrict__ X,        // [T, HID] fp32
    const int*   __restrict__ counts,   // [NEXP]
    const float* __restrict__ Wgu,      // [NEXP, 2*INTER, HID] fp32
    u16*         __restrict__ Hws)      // [T, INTER] bf16 (workspace)
{
  __shared__ __align__(16) u16 sA[BM * LDSP];
  __shared__ __align__(16) u16 sG[BN * LDSP];
  __shared__ __align__(16) u16 sU[BN * LDSP];

  const int n0   = blockIdx.x * BN;
  const int tok0 = blockIdx.y * BM;
  const int e    = find_expert(counts, tok0);

  const float* Wg = Wgu + (size_t)e * (2 * INTER) * HID + (size_t)n0 * HID;
  const float* Wu = Wg + (size_t)INTER * HID;

  const int tid   = threadIdx.x;
  const int lane  = tid & 31;
  const int wid   = tid >> 5;     // 0..7
  const int wm    = wid & 3;      // 4 waves along M (32 rows each)
  const int wn    = wid >> 2;     // 2 waves along N (32 cols each)
  const int lrow  = lane & 15;
  const int lhalf = lane >> 4;

  const int arow = tid >> 1;            // 0..127
  const int akk  = (tid & 1) * 32;      // 0 or 32 (ushorts/floats)
  const int bn   = tid >> 2;            // 0..63
  const int bkk  = (tid & 3) * 16;      // 0,16,32,48

  v8f accG[2][2] = {};
  v8f accU[2][2] = {};

  for (int k0 = 0; k0 < HID; k0 += BK) {
    // --- stage A: X tile [128 x 64] fp32 -> bf16 LDS ---
    {
      const float* s = X + (size_t)(tok0 + arow) * HID + k0 + akk;
      u16* d = &sA[arow * LDSP + akk];
      #pragma unroll
      for (int j = 0; j < 4; ++j) {
        f32x4 x0 = *(const f32x4*)(s + j * 8);
        f32x4 x1 = *(const f32x4*)(s + j * 8 + 4);
        *(u32x4*)(d + j * 8) = pk8(x0, x1);
      }
    }
    // --- stage B: gate + up weight tiles [64 x 64] fp32 -> bf16 LDS ---
    {
      const float* sg = Wg + (size_t)bn * HID + k0 + bkk;
      u16* dg = &sG[bn * LDSP + bkk];
      const float* su = Wu + (size_t)bn * HID + k0 + bkk;
      u16* du = &sU[bn * LDSP + bkk];
      #pragma unroll
      for (int j = 0; j < 2; ++j) {
        f32x4 g0 = *(const f32x4*)(sg + j * 8);
        f32x4 g1 = *(const f32x4*)(sg + j * 8 + 4);
        *(u32x4*)(dg + j * 8) = pk8(g0, g1);
        f32x4 u0 = *(const f32x4*)(su + j * 8);
        f32x4 u1 = *(const f32x4*)(su + j * 8 + 4);
        *(u32x4*)(du + j * 8) = pk8(u0, u1);
      }
      if (k0 + BK < HID) {
        __builtin_prefetch(sg + BK, 0, 0);   // global_prefetch_b8 next K chunk
        __builtin_prefetch(su + BK, 0, 0);
      }
    }
    __syncthreads();

    // --- two 32-deep WMMA groups per stage ---
    #pragma unroll
    for (int kg = 0; kg < 2; ++kg) {
      const int ko = kg * 32;
      Frag a[2], g[2], u[2];
      #pragma unroll
      for (int mf = 0; mf < 2; ++mf) {
        // A 16x32: lane half 0 -> K 0-15 ; half 1 -> K 16-31 (contiguous)
        const u16* p = &sA[(wm * 32 + mf * 16 + lrow) * LDSP + ko + lhalf * 16];
        a[mf].q[0] = *(const u32x4*)(p);
        a[mf].q[1] = *(const u32x4*)(p + 8);
      }
      #pragma unroll
      for (int nf = 0; nf < 2; ++nf) {
        // B 32x16: lane half 0 -> K 0-15 ; half 1 -> K 16-31 (contiguous)
        const u16* pg = &sG[(wn * 32 + nf * 16 + lrow) * LDSP + ko + lhalf * 16];
        g[nf].q[0] = *(const u32x4*)(pg);
        g[nf].q[1] = *(const u32x4*)(pg + 8);
        const u16* pu = &sU[(wn * 32 + nf * 16 + lrow) * LDSP + ko + lhalf * 16];
        u[nf].q[0] = *(const u32x4*)(pu);
        u[nf].q[1] = *(const u32x4*)(pu + 8);
      }
      #pragma unroll
      for (int mf = 0; mf < 2; ++mf)
        #pragma unroll
        for (int nf = 0; nf < 2; ++nf) {
          accG[mf][nf] = __builtin_amdgcn_wmma_f32_16x16x32_bf16(
              false, a[mf].v, false, g[nf].v, (short)0, accG[mf][nf], false, false);
          accU[mf][nf] = __builtin_amdgcn_wmma_f32_16x16x32_bf16(
              false, a[mf].v, false, u[nf].v, (short)0, accU[mf][nf], false, false);
        }
    }
    __syncthreads();
  }

  // --- epilogue: h = silu(gate) * up -> bf16 workspace ---
  #pragma unroll
  for (int mf = 0; mf < 2; ++mf) {
    #pragma unroll
    for (int nf = 0; nf < 2; ++nf) {
      const int col = n0 + wn * 32 + nf * 16 + lrow;
      #pragma unroll
      for (int i = 0; i < 8; ++i) {
        const int row = tok0 + wm * 32 + mf * 16 + lhalf * 8 + i;
        float gg = accG[mf][nf][i];
        float uu = accU[mf][nf][i];
        float hv = (gg / (1.0f + __expf(-gg))) * uu;   // silu(g) * u
        Hws[(size_t)row * INTER + col] = f2bf(hv);
      }
    }
  }
}

// ---------------------------------------------------------------------------
// Kernel 2: Out[T, HID] = h @ Wd^T   per expert segment
// ---------------------------------------------------------------------------
__global__ __launch_bounds__(256) void moe_down(
    const u16*   __restrict__ Hws,      // [T, INTER] bf16
    const int*   __restrict__ counts,   // [NEXP]
    const float* __restrict__ Wd,       // [NEXP, HID, INTER] fp32
    float*       __restrict__ Out)      // [T, HID] fp32
{
  __shared__ __align__(16) u16 sA[BM * LDSP];
  __shared__ __align__(16) u16 sB[BN * LDSP];

  const int n0   = blockIdx.x * BN;
  const int tok0 = blockIdx.y * BM;
  const int e    = find_expert(counts, tok0);
  const float* W = Wd + (size_t)e * HID * INTER + (size_t)n0 * INTER;

  const int tid   = threadIdx.x;
  const int lane  = tid & 31;
  const int wid   = tid >> 5;
  const int wm    = wid & 3;
  const int wn    = wid >> 2;
  const int lrow  = lane & 15;
  const int lhalf = lane >> 4;

  const int arow = tid >> 1;
  const int akk  = (tid & 1) * 32;
  const int bn   = tid >> 2;
  const int bkk  = (tid & 3) * 16;

  v8f acc[2][2] = {};

  for (int k0 = 0; k0 < INTER; k0 += BK) {
    // --- stage A: h tile already bf16 -> async copy straight into LDS ---
    {
      const u16* s = Hws + (size_t)(tok0 + arow) * INTER + k0 + akk;
      u16* d = &sA[arow * LDSP + akk];
#ifdef HAVE_ASYNC_LDS
      #pragma unroll
      for (int j = 0; j < 4; ++j)
        __builtin_amdgcn_global_load_async_to_lds_b128(
            (gas_i32x4*)(s + j * 8),   // int4 __device__ *  (AS1)
            (las_i32x4*)(d + j * 8),   // int4 __shared__ *  (AS3)
            0, 0);
#else
      #pragma unroll
      for (int j = 0; j < 4; ++j)
        *(u32x4*)(d + j * 8) = *(const u32x4*)(s + j * 8);
#endif
    }
    // --- stage B: down weights fp32 -> bf16 ---
    {
      const float* sb = W + (size_t)bn * INTER + k0 + bkk;
      u16* db = &sB[bn * LDSP + bkk];
      #pragma unroll
      for (int j = 0; j < 2; ++j) {
        f32x4 b0 = *(const f32x4*)(sb + j * 8);
        f32x4 b1 = *(const f32x4*)(sb + j * 8 + 4);
        *(u32x4*)(db + j * 8) = pk8(b0, b1);
      }
      if (k0 + BK < INTER) __builtin_prefetch(sb + BK, 0, 0);
    }
#ifdef HAVE_ASYNC_LDS
    __builtin_amdgcn_s_wait_asynccnt(0);   // own async LDS writes visible
#endif
    __syncthreads();

    #pragma unroll
    for (int kg = 0; kg < 2; ++kg) {
      const int ko = kg * 32;
      Frag a[2], b[2];
      #pragma unroll
      for (int mf = 0; mf < 2; ++mf) {
        const u16* p = &sA[(wm * 32 + mf * 16 + lrow) * LDSP + ko + lhalf * 16];
        a[mf].q[0] = *(const u32x4*)(p);
        a[mf].q[1] = *(const u32x4*)(p + 8);
      }
      #pragma unroll
      for (int nf = 0; nf < 2; ++nf) {
        const u16* pb = &sB[(wn * 32 + nf * 16 + lrow) * LDSP + ko + lhalf * 16];
        b[nf].q[0] = *(const u32x4*)(pb);
        b[nf].q[1] = *(const u32x4*)(pb + 8);
      }
      #pragma unroll
      for (int mf = 0; mf < 2; ++mf)
        #pragma unroll
        for (int nf = 0; nf < 2; ++nf)
          acc[mf][nf] = __builtin_amdgcn_wmma_f32_16x16x32_bf16(
              false, a[mf].v, false, b[nf].v, (short)0, acc[mf][nf], false, false);
    }
    __syncthreads();
  }

  #pragma unroll
  for (int mf = 0; mf < 2; ++mf) {
    #pragma unroll
    for (int nf = 0; nf < 2; ++nf) {
      const int col = n0 + wn * 32 + nf * 16 + lrow;
      #pragma unroll
      for (int i = 0; i < 8; ++i) {
        const int row = tok0 + wm * 32 + mf * 16 + lhalf * 8 + i;
        Out[(size_t)row * HID + col] = acc[mf][nf][i];
      }
    }
  }
}

// ---------------------------------------------------------------------------

extern "C" void kernel_launch(void* const* d_in, const int* in_sizes, int n_in,
                              void* d_out, int out_size, void* d_ws, size_t ws_size,
                              hipStream_t stream) {
  (void)in_sizes; (void)n_in; (void)out_size; (void)ws_size;
  const float* X      = (const float*)d_in[0];   // routed_input [T, H]
  const int*   counts = (const int*)d_in[1];     // [NEXP]
  const float* Wgu    = (const float*)d_in[2];   // [NEXP, 2I, H]
  const float* Wd     = (const float*)d_in[3];   // [NEXP, H, I]
  float*       Out    = (float*)d_out;           // [T, H]
  u16*         Hws    = (u16*)d_ws;              // [T, I] bf16 (5.77 MB)

  dim3 blk(256);
  dim3 g1(INTER / BN, TOKENS / BM);   // 22 x 16
  dim3 g2(HID / BN,   TOKENS / BM);   // 16 x 16
  moe_gateup_silu<<<g1, blk, 0, stream>>>(X, counts, Wgu, Hws);
  moe_down<<<g2, blk, 0, stream>>>(Hws, counts, Wd, Out);
}
